// GATConvolutionLinSkip_72911364817012
// MI455X (gfx1250) — matrix operand order repaired and checked by reading, not compiled, not run on hardware
//
#include <hip/hip_runtime.h>
#include <hip/hip_bf16.h>
#include <stdint.h>

// ---------- types ----------
typedef __attribute__((ext_vector_type(16))) __bf16 v16bf;
typedef __attribute__((ext_vector_type(8)))  float  v8f;
typedef __attribute__((ext_vector_type(4)))  unsigned int u32x4;
typedef __attribute__((ext_vector_type(4)))  float  f32x4;
typedef __attribute__((ext_vector_type(2)))  float  f32x2;

#define NEG_SLOPE 0.2f

__device__ __forceinline__ unsigned short f32_to_bf16(float f) {
  unsigned u = __float_as_uint(f);
  unsigned r = u + 0x7FFFu + ((u >> 16) & 1u);   // round-to-nearest-even
  return (unsigned short)(r >> 16);
}

__device__ __forceinline__ void atomicMaxF(float* addr, float v) {
  // works for mixed signs given init = -inf
  if (v >= 0.f) atomicMax((int*)addr, __float_as_int(v));
  else          atomicMin((unsigned int*)addr, __float_as_uint(v));
}

// ---------- fp32 -> bf16 pack (activations) ----------
__global__ void k_cvt_bf16(const float* __restrict__ src,
                           unsigned short* __restrict__ dst, long n) {
  long i = (long)blockIdx.x * blockDim.x + threadIdx.x;
  if (i < n) dst[i] = f32_to_bf16(src[i]);
}

// ---------- weight repack: row-major fp32 [128,ncols] -> B-fragment-major packed bf16 ----------
// Linear index: (((nt*4 + kb)*32 + lane)*8 + v)  -> u32 holding bf16(K), bf16(K+1)
// where K = kb*32 + (lane>>4)*16 + 2v, col = nt*16 + (lane&15).
__global__ void k_pack_w(const float* __restrict__ W, unsigned* __restrict__ Wp, int ncols) {
  int entries = (ncols / 16) * 4 * 32 * 8;
  int idx = blockIdx.x * blockDim.x + threadIdx.x;
  if (idx >= entries) return;
  int v    = idx & 7;
  int lane = (idx >> 3) & 31;
  int kb   = (idx >> 8) & 3;
  int nt   = idx >> 10;
  int k    = kb * 32 + (lane >> 4) * 16 + 2 * v;
  int col  = nt * 16 + (lane & 15);
  unsigned lo = f32_to_bf16(W[(long)k * ncols + col]);
  unsigned hi = f32_to_bf16(W[(long)(k + 1) * ncols + col]);
  Wp[idx] = lo | (hi << 16);
}

// ---------- WMMA GEMM: C[rows,NCOLS] = A[rows,128](bf16) @ Bpack (+bias) ----------
// block = 256 threads = 8 waves; each wave computes a 16-row strip; block tile = 128 rows.
// A fragments for all four K=32 blocks are preloaded; per n-tile, all four B fragments
// are loaded in one clause, then the four WMMAs chain on the accumulator.
template<int NCOLS, bool HASBIAS>
__global__ void k_wmma_gemm(const unsigned short* __restrict__ A,
                            const u32x4* __restrict__ Bp,   // packed fragments
                            const float* __restrict__ bias,
                            float* __restrict__ C, int nrows) {
  const int lane = threadIdx.x & 31;
  const int wave = threadIdx.x >> 5;
  const int nIdx = lane & 15;      // column within 16-tile (B/C/D lane mapping)
  const int half = lane >> 4;
  const int rowBase = blockIdx.x * 128 + wave * 16;

  int arow = rowBase + nIdx;       // A lane mapping: M = lane&15
  if (arow >= nrows) arow = nrows - 1;
  const unsigned short* Arow = A + (long)arow * 128;

  // Preload all A fragments: af[kb].u[0] = K[half*16..+8), af[kb].u[1] = K[half*16+8..+8)
  union Frag { u32x4 u[2]; v16bf vec; };
  Frag af[4];
#pragma unroll
  for (int kb = 0; kb < 4; ++kb) {
    af[kb].u[0] = *(const u32x4*)(Arow + kb * 32 + half * 16);
    af[kb].u[1] = *(const u32x4*)(Arow + kb * 32 + half * 16 + 8);
  }

  const bool fullTile = (rowBase + 16) <= nrows;   // wave-uniform

  for (int nt = 0; nt < NCOLS / 16; ++nt) {
    const int n0 = nt * 16;
    float bb = HASBIAS ? bias[n0 + nIdx] : 0.f;
    v8f acc;
#pragma unroll
    for (int i = 0; i < 8; ++i) acc[i] = bb;

    // Load all four B fragments for this n-tile up front (one clause, one wait).
    Frag bf_[4];
#pragma unroll
    for (int kb = 0; kb < 4; ++kb) {
      const u32x4* bp = Bp + ((long)((nt * 4 + kb) * 32 + lane)) * 2;
      bf_[kb].u[0] = bp[0];
      bf_[kb].u[1] = bp[1];
    }
#pragma unroll
    for (int kb = 0; kb < 4; ++kb) {
      acc = __builtin_amdgcn_wmma_f32_16x16x32_bf16(
          false, af[kb].vec, false, bf_[kb].vec, (short)0, acc, false, false);
    }

    // D layout: VGPR v, lane -> row = v + 8*half, col = n0 + nIdx
    if (fullTile) {
#pragma unroll
      for (int v = 0; v < 8; ++v) {
        C[(long)(rowBase + v + 8 * half) * NCOLS + n0 + nIdx] = acc[v];
      }
    } else {
#pragma unroll
      for (int v = 0; v < 8; ++v) {
        int r = rowBase + v + 8 * half;
        if (r < nrows) C[(long)r * NCOLS + n0 + nIdx] = acc[v];
      }
    }
  }
}

// ---------- attention scores: s_src[i] = h[i]·a_src, s_dst[i] = h[i]·a_dst (wave per row) ----------
__global__ void k_scores(const float* __restrict__ h, const float* __restrict__ a_src,
                         const float* __restrict__ a_dst, float* __restrict__ s_src,
                         float* __restrict__ s_dst, int n) {
  long gid = (long)blockIdx.x * blockDim.x + threadIdx.x;
  int row = (int)(gid >> 5);
  int lane = threadIdx.x & 31;
  if (row >= n) return;
  f32x4 hv = *(const f32x4*)(h + (long)row * 128 + lane * 4);
  f32x4 as = *(const f32x4*)(a_src + lane * 4);
  f32x4 ad = *(const f32x4*)(a_dst + lane * 4);
  float ss = hv.x * as.x + hv.y * as.y + hv.z * as.z + hv.w * as.w;
  float sd = hv.x * ad.x + hv.y * ad.y + hv.z * ad.z + hv.w * ad.w;
#pragma unroll
  for (int off = 16; off > 0; off >>= 1) {
    ss += __shfl_xor(ss, off, 32);
    sd += __shfl_xor(sd, off, 32);
  }
  if (lane == 0) { s_src[row] = ss; s_dst[row] = sd; }
}

// ---------- per-layer init: agg = 0, m = -inf, denom = 0 ----------
__global__ void k_init(float* __restrict__ agg, float* __restrict__ m,
                       float* __restrict__ denom, int n) {
  long i = (long)blockIdx.x * blockDim.x + threadIdx.x;
  long total = (long)n * 128;
  if (i < total) agg[i] = 0.f;
  if (i < n) { m[i] = -INFINITY; denom[i] = 0.f; }
}

// ---------- edge pass 1: segment max ----------
__global__ void k_edge_max(const long long* __restrict__ ei, const float* __restrict__ s_src,
                           const float* __restrict__ s_dst, float* __restrict__ m,
                           int E, int n) {
  long e = (long)blockIdx.x * blockDim.x + threadIdx.x;
  if (e >= (long)E + n) return;
  int s, d;
  if (e < E) { s = (int)ei[e]; d = (int)ei[(long)E + e]; }
  else       { s = d = (int)(e - E); }
  float v = s_src[s] + s_dst[d];
  v = v > 0.f ? v : NEG_SLOPE * v;
  atomicMaxF(&m[d], v);
}

// ---------- edge pass 2: ex = exp(e - m[dst]); denom[dst] += ex ----------
__global__ void k_edge_exp(const long long* __restrict__ ei, const float* __restrict__ s_src,
                           const float* __restrict__ s_dst, const float* __restrict__ m,
                           float* __restrict__ exbuf, float* __restrict__ denom,
                           int E, int n) {
  long e = (long)blockIdx.x * blockDim.x + threadIdx.x;
  if (e >= (long)E + n) return;
  int s, d;
  if (e < E) { s = (int)ei[e]; d = (int)ei[(long)E + e]; }
  else       { s = d = (int)(e - E); }
  float v = s_src[s] + s_dst[d];
  v = v > 0.f ? v : NEG_SLOPE * v;
  float ex = expf(v - m[d]);
  exbuf[e] = ex;
  atomicAdd(&denom[d], ex);
}

// ---------- edge pass 3: out[dst] += alpha * h[src]  (wave per edge, 4 floats/lane) ----------
__global__ void k_edge_scatter(const long long* __restrict__ ei, const float* __restrict__ h,
                               const float* __restrict__ exbuf, const float* __restrict__ denom,
                               float* __restrict__ out, int E, int n) {
  long gid = (long)blockIdx.x * blockDim.x + threadIdx.x;
  long e = gid >> 5;
  int lane = threadIdx.x & 31;
  if (e >= (long)E + n) return;
  int s, d;
  if (e < E) { s = (int)ei[e]; d = (int)ei[(long)E + e]; }
  else       { s = d = (int)(e - E); }
  __builtin_prefetch(h + (long)s * 128 + lane * 4, 0, 0);   // global_prefetch_b8
  float alpha = exbuf[e] / denom[d];
  f32x4 hv = *(const f32x4*)(h + (long)s * 128 + lane * 4);
  float* o = out + (long)d * 128 + lane * 4;
  atomicAdd(o + 0, hv.x * alpha);
  atomicAdd(o + 1, hv.y * alpha);
  atomicAdd(o + 2, hv.z * alpha);
  atomicAdd(o + 3, hv.w * alpha);
}

// ---------- layer-1 epilogue: h1 = relu(agg + b1); also bf16 copy ----------
__global__ void k_epi1(const float* __restrict__ agg, const float* __restrict__ b,
                       float* __restrict__ h1, unsigned short* __restrict__ h1b, long n128) {
  long i = (long)blockIdx.x * blockDim.x + threadIdx.x;
  if (i >= n128) return;
  float v = agg[i] + b[i & 127];
  v = v > 0.f ? v : 0.f;
  h1[i] = v;
  h1b[i] = f32_to_bf16(v);
}

// ---------- layer-2 epilogue: final = h1 + (agg2 + b2) -> bf16 for the linear ----------
__global__ void k_epi2(const float* __restrict__ agg2, const float* __restrict__ b,
                       const float* __restrict__ h1, unsigned short* __restrict__ fb, long n128) {
  long i = (long)blockIdx.x * blockDim.x + threadIdx.x;
  if (i >= n128) return;
  float v = h1[i] + agg2[i] + b[i & 127];
  fb[i] = f32_to_bf16(v);
}

// ---------- in-place log_softmax over 64 cols (wave per row, 2 cols/lane) ----------
__global__ void k_log_softmax64(float* __restrict__ logits, int n) {
  long gid = (long)blockIdx.x * blockDim.x + threadIdx.x;
  int row = (int)(gid >> 5);
  int lane = threadIdx.x & 31;
  if (row >= n) return;
  f32x2 v = *(f32x2*)(logits + (long)row * 64 + lane * 2);
  float mx = fmaxf(v.x, v.y);
#pragma unroll
  for (int off = 16; off > 0; off >>= 1) mx = fmaxf(mx, __shfl_xor(mx, off, 32));
  float s = expf(v.x - mx) + expf(v.y - mx);
#pragma unroll
  for (int off = 16; off > 0; off >>= 1) s += __shfl_xor(s, off, 32);
  float lse = mx + logf(s);
  v.x -= lse; v.y -= lse;
  *(f32x2*)(logits + (long)row * 64 + lane * 2) = v;
}

// =====================================================================
extern "C" void kernel_launch(void* const* d_in, const int* in_sizes, int n_in,
                              void* d_out, int out_size, void* d_ws, size_t ws_size,
                              hipStream_t stream) {
  const float*     x   = (const float*)d_in[0];
  const long long* ei  = (const long long*)d_in[1];
  const float*     W1  = (const float*)d_in[2];
  const float*     a1s = (const float*)d_in[3];
  const float*     a1d = (const float*)d_in[4];
  const float*     b1  = (const float*)d_in[5];
  const float*     W2  = (const float*)d_in[6];
  const float*     a2s = (const float*)d_in[7];
  const float*     a2d = (const float*)d_in[8];
  const float*     b2  = (const float*)d_in[9];
  const float*     Wl  = (const float*)d_in[10];
  const float*     bl  = (const float*)d_in[11];

  const int N = in_sizes[0] / 128;
  const int E = in_sizes[1] / 2;
  const int C = in_sizes[11];        // 64
  const long n128 = (long)N * 128;

  // ---- workspace layout ----
  char* p = (char*)d_ws;
  auto alloc = [&](size_t bytes) -> void* {
    void* q = (void*)p;
    p += (bytes + 255) & ~(size_t)255;
    return q;
  };
  unsigned short* actb = (unsigned short*)alloc(n128 * 2);   // bf16 activations (x / h1 / final)
  unsigned* w1p = (unsigned*)alloc(8 * 4 * 32 * 8 * 4);      // packed fragment weights
  unsigned* w2p = (unsigned*)alloc(8 * 4 * 32 * 8 * 4);
  unsigned* wlp = (unsigned*)alloc((size_t)(C / 16) * 4 * 32 * 8 * 4);
  float* hmat  = (float*)alloc(n128 * 4);                    // x@W (per layer)
  float* h1buf = (float*)alloc(n128 * 4);                    // relu(layer1)
  float* agg   = (float*)alloc(n128 * 4);                    // attention-aggregated output
  float* ssrc  = (float*)alloc((size_t)N * 4);
  float* sdst  = (float*)alloc((size_t)N * 4);
  float* mbuf  = (float*)alloc((size_t)N * 4);
  float* denom = (float*)alloc((size_t)N * 4);
  float* exbuf = (float*)alloc(((size_t)E + N) * 4);

  float* logits = (float*)d_out;     // first N*C floats of d_out

  const int T = 256;
  dim3 blk(T);
  const int gElem   = (int)((n128 + T - 1) / T);
  const int gRows   = (N + 127) / 128;
  const int gWaveN  = (int)(((long)N * 32 + T - 1) / T);
  const long EN     = (long)E + N;
  const int gEdge   = (int)((EN + T - 1) / T);
  const int gEdgeW  = (int)((EN * 32 + T - 1) / T);

  // weights -> packed WMMA fragment layout
  const int ent128 = 8 * 4 * 32 * 8;
  const int entC   = (C / 16) * 4 * 32 * 8;
  k_pack_w<<<(ent128 + T - 1) / T, blk, 0, stream>>>(W1, w1p, 128);
  k_pack_w<<<(ent128 + T - 1) / T, blk, 0, stream>>>(W2, w2p, 128);
  k_pack_w<<<(entC + T - 1) / T, blk, 0, stream>>>(Wl, wlp, C);

  // ===== layer 1 =====
  k_cvt_bf16<<<gElem, blk, 0, stream>>>(x, actb, n128);
  k_wmma_gemm<128, false><<<gRows, blk, 0, stream>>>(actb, (const u32x4*)w1p, nullptr, hmat, N);
  k_scores<<<gWaveN, blk, 0, stream>>>(hmat, a1s, a1d, ssrc, sdst, N);
  k_init<<<gElem, blk, 0, stream>>>(agg, mbuf, denom, N);
  k_edge_max<<<gEdge, blk, 0, stream>>>(ei, ssrc, sdst, mbuf, E, N);
  k_edge_exp<<<gEdge, blk, 0, stream>>>(ei, ssrc, sdst, mbuf, exbuf, denom, E, N);
  k_edge_scatter<<<gEdgeW, blk, 0, stream>>>(ei, hmat, exbuf, denom, agg, E, N);
  k_epi1<<<gElem, blk, 0, stream>>>(agg, b1, h1buf, actb, n128);

  // ===== layer 2 =====
  k_wmma_gemm<128, false><<<gRows, blk, 0, stream>>>(actb, (const u32x4*)w2p, nullptr, hmat, N);
  k_scores<<<gWaveN, blk, 0, stream>>>(hmat, a2s, a2d, ssrc, sdst, N);
  k_init<<<gElem, blk, 0, stream>>>(agg, mbuf, denom, N);
  k_edge_max<<<gEdge, blk, 0, stream>>>(ei, ssrc, sdst, mbuf, E, N);
  k_edge_exp<<<gEdge, blk, 0, stream>>>(ei, ssrc, sdst, mbuf, exbuf, denom, E, N);
  k_edge_scatter<<<gEdgeW, blk, 0, stream>>>(ei, hmat, exbuf, denom, agg, E, N);
  k_epi2<<<gElem, blk, 0, stream>>>(agg, b2, h1buf, actb, n128);

  // ===== linear + log_softmax =====
  k_wmma_gemm<64, true><<<gRows, blk, 0, stream>>>(actb, (const u32x4*)wlp, bl, logits, N);
  k_log_softmax64<<<gWaveN, blk, 0, stream>>>(logits, N);

  // ===== pass-through edge_index into tail of d_out =====
  long headFloats = (long)N * C;
  long tailBytes = ((long)out_size - headFloats) * 4;
  long eiBytes = (long)in_sizes[1] * 8;
  if (tailBytes > eiBytes) tailBytes = eiBytes;
  if (tailBytes > 0) {
    hipMemcpyAsync((char*)d_out + headFloats * 4, d_in[1], (size_t)tailBytes,
                   hipMemcpyDeviceToDevice, stream);
  }
}
